// TransformerBlock_58059367907433
// MI455X (gfx1250) — compile-verified, matrix operand order checked by reading, not run on hardware
//
#include <hip/hip_runtime.h>

// ---------------------------------------------------------------------------
// Transformer block for MI455X (gfx1250, wave32, WMMA + TDM).
// bf16 data / f32 accumulate via v_wmma_f32_16x16x32_bf16.
// GEMM tiles staged into LDS by the Tensor Data Mover (tensor_load_to_lds)
// when the builtin is available; manual cooperative copy otherwise.
// ---------------------------------------------------------------------------

typedef __bf16 bf16_t;
typedef __attribute__((ext_vector_type(16))) __bf16        v16bf;
typedef __attribute__((ext_vector_type(8)))  float         v8f;
typedef __attribute__((ext_vector_type(4)))  unsigned int  v4u;
typedef __attribute__((ext_vector_type(4)))  int           v4i;
typedef __attribute__((ext_vector_type(8)))  int           v8i;

#define D_MODEL 1024
#define SEQ     2048
#define BATCH   2
#define HEADS   16
#define DK      64
#define D_FF    4096
#define M_ROWS  (BATCH * SEQ)   // 4096

#if defined(__has_builtin)
#if __has_builtin(__builtin_amdgcn_tensor_load_to_lds) && \
    __has_builtin(__builtin_amdgcn_s_wait_tensorcnt)
#define USE_TDM 1
#endif
#endif

union FragU { v16bf v; uint4 q[2]; };

__device__ inline v16bf ld_frag16(const bf16_t* p0, const bf16_t* p1) {
  FragU u;
  u.q[0] = *reinterpret_cast<const uint4*>(p0);
  u.q[1] = *reinterpret_cast<const uint4*>(p1);
  return u.v;
}

__device__ inline v8f wmma_bf16(v16bf a, v16bf b, v8f c) {
  // (neg_a, A, neg_b, B, c_mod, C, reuse_a, reuse_b)
  return __builtin_amdgcn_wmma_f32_16x16x32_bf16(false, a, false, b, (short)0, c,
                                                 false, false);
}

#ifdef USE_TDM
// ---------------------------------------------------------------------------
// TDM 2D tile load: 128 rows x 32 bf16 (64B) per row, global -> LDS, with
// LDS padding of 4 DWORDs after every 16 DWORDs stored (=> 80B row pitch,
// matching the [128][40] bf16 LDS tiles used by the WMMA fragment loads).
// D# layout per CDNA5 ISA 8.3/8.4 (group0 128b, group1 256b, groups2/3 zero).
// ---------------------------------------------------------------------------
__device__ inline void tdm_load_tile_128x32(const bf16_t* gtile,
                                            unsigned lds_off,
                                            unsigned k_len,   // tensor_dim0
                                            unsigned rows) {  // tensor_dim1
  const unsigned long long ga = (unsigned long long)(size_t)gtile;
  v4u g0;
  g0[0] = 1u;                                   // count=1, user desc
  g0[1] = lds_off;                              // lds_addr (bytes)
  g0[2] = (unsigned)ga;                         // global_addr[31:0]
  g0[3] = (unsigned)((ga >> 32) & 0x01FFFFFFull) | (2u << 30);  // [56:32]|type=2
  v8i g1;
  // data_size=1 (2B), pad_enable=1, pad_interval=3 (16 DW), pad_amount=3 (4 DW)
  g1[0] = (int)((1u << 16) | (1u << 20) | (3u << 22) | (3u << 25));
  g1[1] = (int)((k_len & 0xFFFFu) << 16);                       // dim0[15:0]
  g1[2] = (int)(((k_len >> 16) & 0xFFFFu) | ((rows & 0xFFFFu) << 16));
  g1[3] = (int)(((rows >> 16) & 0xFFFFu) | (32u << 16));        // tile_dim0=32
  g1[4] = (int)128u;                            // tile_dim1=128, tile_dim2=0
  g1[5] = (int)k_len;                           // tensor_dim0_stride lo32
  g1[6] = 0;                                    // stride0 hi16 | stride1 lo16
  g1[7] = 0;                                    // stride1 hi32
  v4i g2 = {0, 0, 0, 0};
  v4i g3 = {0, 0, 0, 0};
#if __clang_major__ >= 23
  v8i g4 = {0, 0, 0, 0, 0, 0, 0, 0};
  __builtin_amdgcn_tensor_load_to_lds(g0, g1, g2, g3, g4, 0);
#else
  __builtin_amdgcn_tensor_load_to_lds(g0, g1, g2, g3, 0);
#endif
}
#endif  // USE_TDM

// ---------------------------------------------------------------------------
// Weight transpose + fp32 -> bf16 convert:  W[K][N] -> Wt[N][K]
// ---------------------------------------------------------------------------
__global__ __launch_bounds__(256) void transpose_to_bf16(
    const float* __restrict__ W, bf16_t* __restrict__ Wt, int K, int N) {
  __shared__ float tile[32][33];
  const int tx = threadIdx.x & 31, ty = threadIdx.x >> 5;  // 32 x 8
  const int n0 = blockIdx.x * 32, k0 = blockIdx.y * 32;
#pragma unroll
  for (int i = 0; i < 4; ++i)
    tile[ty + i * 8][tx] = W[(size_t)(k0 + ty + i * 8) * N + n0 + tx];
  __syncthreads();
#pragma unroll
  for (int i = 0; i < 4; ++i)
    Wt[(size_t)(n0 + ty + i * 8) * K + k0 + tx] = (bf16_t)tile[tx][ty + i * 8];
}

// ---------------------------------------------------------------------------
// RMSNorm: fp32 row -> bf16 normalized row (one block per row, D_MODEL=1024)
// ---------------------------------------------------------------------------
__global__ __launch_bounds__(256) void rmsnorm_kernel(
    const float* __restrict__ x, const float* __restrict__ gain,
    bf16_t* __restrict__ out) {
  __shared__ float warp_sum[8];
  const int tid = threadIdx.x;
  const float4 v = reinterpret_cast<const float4*>(x + (size_t)blockIdx.x * D_MODEL)[tid];
  float ss = v.x * v.x + v.y * v.y + v.z * v.z + v.w * v.w;
  ss += __shfl_xor(ss, 1, 32);
  ss += __shfl_xor(ss, 2, 32);
  ss += __shfl_xor(ss, 4, 32);
  ss += __shfl_xor(ss, 8, 32);
  ss += __shfl_xor(ss, 16, 32);
  if ((tid & 31) == 0) warp_sum[tid >> 5] = ss;
  __syncthreads();
  float tot = 0.f;
#pragma unroll
  for (int i = 0; i < 8; ++i) tot += warp_sum[i];
  const float inv = rsqrtf(tot * (1.0f / D_MODEL) + 1e-6f);
  const float4 gv = reinterpret_cast<const float4*>(gain)[tid];
  bf16_t* orow = out + (size_t)blockIdx.x * D_MODEL + tid * 4;
  orow[0] = (bf16_t)(v.x * inv * gv.x);
  orow[1] = (bf16_t)(v.y * inv * gv.y);
  orow[2] = (bf16_t)(v.z * inv * gv.z);
  orow[3] = (bf16_t)(v.w * inv * gv.w);
}

// ---------------------------------------------------------------------------
// GEMM: C[M][N] = A[M][K] (bf16, k-major) x Bt[N][K] (bf16, k-major)
// 256 threads = 8 waves; block tile 128x128; per-wave 32x64 (2x4 frags).
// EPI: 0 = store bf16, 1 = fp32 out = resid + C, 2 = bf16 scatter to V^T
// Assumes M,N % 128 == 0 and K % 32 == 0 (true for all calls here).
// ---------------------------------------------------------------------------
template <int EPI>
__global__ __launch_bounds__(256) void gemm_bf16_wmma(
    const bf16_t* __restrict__ A, const bf16_t* __restrict__ Bt,
    int M, int N, int K, const float* __restrict__ resid,
    void* __restrict__ outp) {
  __shared__ bf16_t sA[128][40];   // pad 8 -> conflict-free b128 frag loads
  __shared__ bf16_t sB[128][40];
  const int tid  = threadIdx.x;
  const int lane = tid & 31, wave = tid >> 5;
  const int g = lane >> 4, ln = lane & 15;
  const int wm = wave >> 1, wn = wave & 1;
  const int m_blk = blockIdx.y * 128, n_blk = blockIdx.x * 128;
#ifdef USE_TDM
  const unsigned ldsA_off = (unsigned)(size_t)(void*)&sA[0][0];  // addr[31:0]
  const unsigned ldsB_off = (unsigned)(size_t)(void*)&sB[0][0];
#else
  const int ldr = tid >> 1;         // 0..127
  const int ldc = (tid & 1) * 16;   // 0 or 16
#endif

  v8f acc[2][4] = {};

  for (int k0 = 0; k0 < K; k0 += 32) {
    __syncthreads();
#ifdef USE_TDM
    if (wave == 0) {
      tdm_load_tile_128x32(A + (size_t)m_blk * K + k0, ldsA_off,
                           (unsigned)K, (unsigned)M);
      tdm_load_tile_128x32(Bt + (size_t)n_blk * K + k0, ldsB_off,
                           (unsigned)K, (unsigned)N);
      __builtin_amdgcn_s_wait_tensorcnt(0);
    }
#else
    {
      const uint4* srcA =
          reinterpret_cast<const uint4*>(A + (size_t)(m_blk + ldr) * K + k0 + ldc);
      uint4* dstA = reinterpret_cast<uint4*>(&sA[ldr][ldc]);
      dstA[0] = srcA[0];
      dstA[1] = srcA[1];
      const uint4* srcB =
          reinterpret_cast<const uint4*>(Bt + (size_t)(n_blk + ldr) * K + k0 + ldc);
      uint4* dstB = reinterpret_cast<uint4*>(&sB[ldr][ldc]);
      dstB[0] = srcB[0];
      dstB[1] = srcB[1];
    }
#endif
    __syncthreads();

    v16bf aF[2], bF[4];
#pragma unroll
    for (int mi = 0; mi < 2; ++mi) {
      const int r = wm * 32 + mi * 16 + ln;
      aF[mi] = ld_frag16(&sA[r][g * 8], &sA[r][16 + g * 8]);
    }
#pragma unroll
    for (int ni = 0; ni < 4; ++ni) {
      const int r = wn * 64 + ni * 16 + ln;
      bF[ni] = ld_frag16(&sB[r][g * 8], &sB[r][16 + g * 8]);
    }
#pragma unroll
    for (int mi = 0; mi < 2; ++mi)
#pragma unroll
      for (int ni = 0; ni < 4; ++ni)
        acc[mi][ni] = wmma_bf16(aF[mi], bF[ni], acc[mi][ni]);
  }

  // Epilogue. D layout: element r -> m = r + 8*(lane/16), n = lane%16.
#pragma unroll
  for (int mi = 0; mi < 2; ++mi)
#pragma unroll
    for (int ni = 0; ni < 4; ++ni)
#pragma unroll
      for (int r = 0; r < 8; ++r) {
        const int mg = m_blk + wm * 32 + mi * 16 + r + 8 * g;
        const int ng = n_blk + wn * 64 + ni * 16 + ln;
        const float c = acc[mi][ni][r];
        if (EPI == 0) {
          ((bf16_t*)outp)[(size_t)mg * N + ng] = (bf16_t)c;
        } else if (EPI == 1) {
          const size_t idx = (size_t)mg * N + ng;
          ((float*)outp)[idx] = resid[idx] + c;
        } else {  // V^T scatter: [B,H,DK,SEQ]
          const int bb = mg >> 11, sin = mg & (SEQ - 1);
          const int hh = ng >> 6, dd = ng & (DK - 1);
          ((bf16_t*)outp)[(((size_t)(bb * HEADS + hh) * DK + dd) << 11) + sin] =
              (bf16_t)c;
        }
      }
}

// ---------------------------------------------------------------------------
// Flash attention (causal). Q,K: bf16 [B*S][D_MODEL]; Vt: bf16 [B,H,DK,SEQ].
// 128 threads = 4 waves; each wave owns 16 query rows; kv tiles of 32.
// Online softmax lives in the WMMA D-fragment layout. Row max uses shfl_xor
// butterflies within 16-lane halves (wave32); the softmax denominator is
// accumulated by an extra P x ones WMMA column (gets the same alpha rescale),
// eliminating all sum reductions.
// ---------------------------------------------------------------------------
__global__ __launch_bounds__(128) void flash_attn_wmma(
    const bf16_t* __restrict__ Q, const bf16_t* __restrict__ Km,
    const bf16_t* __restrict__ Vt, bf16_t* __restrict__ O) {
  __shared__ bf16_t sP[4][16][40];  // per-wave P staging (D-layout -> A-layout)
  const int tid  = threadIdx.x;
  const int lane = tid & 31, wave = tid >> 5;
  const int g = lane >> 4, ln = lane & 15;
  const int bh = blockIdx.y, b = bh >> 4, h = bh & 15;
  const int q0 = blockIdx.x * 64 + wave * 16;

  // Q A-fragments: 16 rows x 64 (two k-halves of 32)
  const bf16_t* qrow = Q + (size_t)(b * SEQ + q0 + ln) * D_MODEL + h * DK;
  v16bf qf[2];
#pragma unroll
  for (int kk = 0; kk < 2; ++kk)
    qf[kk] = ld_frag16(qrow + kk * 32 + g * 8, qrow + kk * 32 + 16 + g * 8);

  v16bf onesf;
#pragma unroll
  for (int j = 0; j < 16; ++j) onesf[j] = (bf16_t)1.0f;

  v8f o[4] = {};
  v8f o5 = {};  // softmax denominator accumulator (P x ones)
  float mrow[8];
#pragma unroll
  for (int r = 0; r < 8; ++r) mrow[r] = -__builtin_inff();

  for (int t0 = 0; t0 <= q0 + 15; t0 += 32) {
    // scores S = Q * K^T  (16 x 32), two 16x16 D frags
    v8f s[2] = {};
#pragma unroll
    for (int ni = 0; ni < 2; ++ni) {
      const bf16_t* krow =
          Km + (size_t)(b * SEQ + t0 + ni * 16 + ln) * D_MODEL + h * DK;
#pragma unroll
      for (int kk = 0; kk < 2; ++kk) {
        v16bf kf = ld_frag16(krow + kk * 32 + g * 8, krow + kk * 32 + 16 + g * 8);
        s[ni] = wmma_bf16(qf[kk], kf, s[ni]);
      }
    }
    // scale + causal mask
#pragma unroll
    for (int ni = 0; ni < 2; ++ni)
#pragma unroll
      for (int r = 0; r < 8; ++r) {
        const float v = s[ni][r] * 0.125f;  // 1/sqrt(64)
        const int qg = q0 + r + 8 * g;
        const int tg = t0 + ni * 16 + ln;
        s[ni][r] = (tg > qg) ? -__builtin_inff() : v;
      }
    // row max (butterfly within 16-lane halves) + online rescale factor
    float alpha[8];
#pragma unroll
    for (int r = 0; r < 8; ++r) {
      float v = fmaxf(s[0][r], s[1][r]);
      v = fmaxf(v, __shfl_xor(v, 1, 32));
      v = fmaxf(v, __shfl_xor(v, 2, 32));
      v = fmaxf(v, __shfl_xor(v, 4, 32));
      v = fmaxf(v, __shfl_xor(v, 8, 32));
      const float mn = fmaxf(mrow[r], v);
      alpha[r] = __expf(mrow[r] - mn);
      mrow[r] = mn;
    }
#pragma unroll
    for (int ni = 0; ni < 2; ++ni)
#pragma unroll
      for (int r = 0; r < 8; ++r) s[ni][r] = __expf(s[ni][r] - mrow[r]);
#pragma unroll
    for (int f = 0; f < 4; ++f)
#pragma unroll
      for (int r = 0; r < 8; ++r) o[f][r] *= alpha[r];
#pragma unroll
    for (int r = 0; r < 8; ++r) o5[r] *= alpha[r];

    // P: D-layout -> bf16 LDS -> A-layout (wave-local, split-counter sync)
#pragma unroll
    for (int ni = 0; ni < 2; ++ni)
#pragma unroll
      for (int r = 0; r < 8; ++r)
        sP[wave][r + 8 * g][ni * 16 + ln] = (bf16_t)s[ni][r];
    asm volatile("s_wait_dscnt 0" ::: "memory");
    v16bf pf = ld_frag16(&sP[wave][ln][g * 8], &sP[wave][ln][16 + g * 8]);

    // O += P(16x32) * V(32x64): B frags from V^T rows (contiguous along t)
#pragma unroll
    for (int f = 0; f < 4; ++f) {
      const bf16_t* vrow =
          Vt + (size_t)((b * HEADS + h) * DK + f * 16 + ln) * SEQ + t0;
      v16bf vf = ld_frag16(vrow + g * 8, vrow + 16 + g * 8);
      o[f] = wmma_bf16(pf, vf, o[f]);
    }
    // denominator: l += P x ones
    o5 = wmma_bf16(pf, onesf, o5);
  }

  // normalize and store: out[b, q, h*64 + d] (bf16, row-major D_MODEL)
#pragma unroll
  for (int f = 0; f < 4; ++f)
#pragma unroll
    for (int r = 0; r < 8; ++r) {
      const float val = o[f][r] / o5[r];
      const int qg = q0 + r + 8 * g;
      O[(size_t)(b * SEQ + qg) * D_MODEL + h * DK + f * 16 + ln] = (bf16_t)val;
    }
}

// ---------------------------------------------------------------------------
// SwiGLU elementwise: gate = silu(gate) * up   (bf16, in-place)
// ---------------------------------------------------------------------------
__global__ __launch_bounds__(256) void swiglu_kernel(
    bf16_t* __restrict__ gate, const bf16_t* __restrict__ up, int n) {
  const int i4 = (blockIdx.x * 256 + threadIdx.x) * 4;
  if (i4 >= n) return;
#pragma unroll
  for (int j = 0; j < 4; ++j) {
    const float gv = (float)gate[i4 + j];
    const float uv = (float)up[i4 + j];
    const float sil = gv / (1.f + __expf(-gv));
    gate[i4 + j] = (bf16_t)(sil * uv);
  }
}

// ---------------------------------------------------------------------------
// Host orchestration
// ---------------------------------------------------------------------------
extern "C" void kernel_launch(void* const* d_in, const int* in_sizes, int n_in,
                              void* d_out, int out_size, void* d_ws,
                              size_t ws_size, hipStream_t stream) {
  (void)in_sizes; (void)n_in; (void)out_size; (void)ws_size;
  const float* x      = (const float*)d_in[0];
  const float* w_q    = (const float*)d_in[1];
  const float* w_k    = (const float*)d_in[2];
  const float* w_v    = (const float*)d_in[3];
  const float* w_o    = (const float*)d_in[4];
  const float* w_gate = (const float*)d_in[5];
  const float* w_up   = (const float*)d_in[6];
  const float* w_down = (const float*)d_in[7];
  const float* g1     = (const float*)d_in[8];
  const float* g2     = (const float*)d_in[9];

  unsigned char* ws = (unsigned char*)d_ws;
  const size_t MB = 1ull << 20;
  bf16_t* wtq  = (bf16_t*)(ws + 0 * MB);    // [1024][1024]      2 MB
  bf16_t* wtk  = (bf16_t*)(ws + 2 * MB);    //                   2 MB
  bf16_t* wtv  = (bf16_t*)(ws + 4 * MB);    //                   2 MB
  bf16_t* wto  = (bf16_t*)(ws + 6 * MB);    //                   2 MB
  bf16_t* wtg  = (bf16_t*)(ws + 8 * MB);    // [4096][1024]      8 MB
  bf16_t* wtu  = (bf16_t*)(ws + 16 * MB);   //                   8 MB
  bf16_t* wtd  = (bf16_t*)(ws + 24 * MB);   // [1024][4096]      8 MB
  bf16_t* nx1  = (bf16_t*)(ws + 32 * MB);   // [4096][1024]      8 MB
  bf16_t* qb   = (bf16_t*)(ws + 40 * MB);   // [4096][1024]      8 MB
  bf16_t* kb   = (bf16_t*)(ws + 48 * MB);   //                   8 MB
  bf16_t* vtb  = (bf16_t*)(ws + 56 * MB);   // [B,H,DK,SEQ]      8 MB
  bf16_t* ao   = (bf16_t*)(ws + 64 * MB);   // attn out          8 MB
  float*  x2   = (float*) (ws + 72 * MB);   // f32 resid        16 MB
  bf16_t* nx2  = (bf16_t*)(ws + 88 * MB);   // [4096][1024]      8 MB
  bf16_t* gate = (bf16_t*)(ws + 96 * MB);   // [4096][4096]     32 MB
  bf16_t* up   = (bf16_t*)(ws + 128 * MB);  //                  32 MB (end 160 MB)

  // --- weight prep: W[K][N] -> Wt[N][K] bf16 ---
  transpose_to_bf16<<<dim3(D_MODEL / 32, D_MODEL / 32), 256, 0, stream>>>(w_q, wtq, D_MODEL, D_MODEL);
  transpose_to_bf16<<<dim3(D_MODEL / 32, D_MODEL / 32), 256, 0, stream>>>(w_k, wtk, D_MODEL, D_MODEL);
  transpose_to_bf16<<<dim3(D_MODEL / 32, D_MODEL / 32), 256, 0, stream>>>(w_v, wtv, D_MODEL, D_MODEL);
  transpose_to_bf16<<<dim3(D_MODEL / 32, D_MODEL / 32), 256, 0, stream>>>(w_o, wto, D_MODEL, D_MODEL);
  transpose_to_bf16<<<dim3(D_FF / 32, D_MODEL / 32), 256, 0, stream>>>(w_gate, wtg, D_MODEL, D_FF);
  transpose_to_bf16<<<dim3(D_FF / 32, D_MODEL / 32), 256, 0, stream>>>(w_up, wtu, D_MODEL, D_FF);
  transpose_to_bf16<<<dim3(D_MODEL / 32, D_FF / 32), 256, 0, stream>>>(w_down, wtd, D_FF, D_MODEL);

  // --- attention sublayer ---
  rmsnorm_kernel<<<M_ROWS, 256, 0, stream>>>(x, g1, nx1);

  const dim3 gemm_blk(256);
  gemm_bf16_wmma<0><<<dim3(D_MODEL / 128, M_ROWS / 128), gemm_blk, 0, stream>>>(
      nx1, wtq, M_ROWS, D_MODEL, D_MODEL, nullptr, qb);
  gemm_bf16_wmma<0><<<dim3(D_MODEL / 128, M_ROWS / 128), gemm_blk, 0, stream>>>(
      nx1, wtk, M_ROWS, D_MODEL, D_MODEL, nullptr, kb);
  gemm_bf16_wmma<2><<<dim3(D_MODEL / 128, M_ROWS / 128), gemm_blk, 0, stream>>>(
      nx1, wtv, M_ROWS, D_MODEL, D_MODEL, nullptr, vtb);

  flash_attn_wmma<<<dim3(SEQ / 64, BATCH * HEADS), 128, 0, stream>>>(qb, kb, vtb, ao);

  gemm_bf16_wmma<1><<<dim3(D_MODEL / 128, M_ROWS / 128), gemm_blk, 0, stream>>>(
      ao, wto, M_ROWS, D_MODEL, D_MODEL, x, x2);

  // --- SwiGLU FFN sublayer ---
  rmsnorm_kernel<<<M_ROWS, 256, 0, stream>>>(x2, g2, nx2);

  gemm_bf16_wmma<0><<<dim3(D_FF / 128, M_ROWS / 128), gemm_blk, 0, stream>>>(
      nx2, wtg, M_ROWS, D_FF, D_MODEL, nullptr, gate);
  gemm_bf16_wmma<0><<<dim3(D_FF / 128, M_ROWS / 128), gemm_blk, 0, stream>>>(
      nx2, wtu, M_ROWS, D_FF, D_MODEL, nullptr, up);

  swiglu_kernel<<<(M_ROWS * D_FF) / 1024, 256, 0, stream>>>(gate, up, M_ROWS * D_FF);

  gemm_bf16_wmma<1><<<dim3(D_MODEL / 128, M_ROWS / 128), gemm_blk, 0, stream>>>(
      gate, wtd, M_ROWS, D_MODEL, D_FF, x2, (float*)d_out);
}
